// EEG_Encoder_49512382988750
// MI455X (gfx1250) — matrix operand order repaired and checked by reading, not compiled, not run on hardware
//
#include <hip/hip_runtime.h>

typedef __attribute__((ext_vector_type(16))) _Float16 v16h;
typedef __attribute__((ext_vector_type(8)))  _Float16 v8h;
typedef __attribute__((ext_vector_type(8)))  float    v8f;

union V16U { v16h v; v8h h2[2]; };

// ---------------- LDS layout (bytes) ----------------
// Hs    f32 [496][40]                 @ 0       79360
// HF    f16 [496][64]  (A panel)      @ 79360   63488
// WVP   f16 packed B-frags 3nt x 2ks  @ 142848   6144
// W1P   f16 packed B-frags 10nt x 2ks @ 148992  20480
// W2P   f16 packed B-frags 3nt x 5ks  @ 169472  15360
// T1    f16 [8 waves][16][160]        @ 184832  40960
// SCR   f32 [8 waves][16][48]         @ 225792  24576
// overlays: XS f32[15000] @142848, CWL/PEL @225792,
//           POOL @148992, PART @142848, LG @169472
#define SMEM_BYTES 250368
#define XS_LDS_OFF 142848u

__device__ __forceinline__ v16h frag_from_chunks(const _Float16* lo, const _Float16* hi) {
  V16U u;
  u.h2[0] = *(const v8h*)lo;
  u.h2[1] = *(const v8h*)hi;
  return u.v;
}

// A fragment: 16xK tile, row-major, stride lda (halfs). Per ISA 16-bit A layout:
// lane m = lane&15, k-base = (lane>>4)*8; elements 0..7 -> k=kb..kb+7,
// elements 8..15 -> k=16+kb..16+kb+7  (two contiguous 16B chunks).
__device__ __forceinline__ v16h load_a(const _Float16* A, int lda, int koff, int lane) {
  int m  = lane & 15;
  int kb = (lane >> 4) << 3;
  const _Float16* r = A + m * lda + koff + kb;
  return frag_from_chunks(r, r + 16);
}

// B fragment from a pre-permuted packed panel: contiguous 32B per lane.
__device__ __forceinline__ v16h load_bp(const _Float16* P, int frag_idx, int lane) {
  return *(const v16h*)(P + (size_t)(frag_idx * 32 + lane) * 16);
}

__device__ __forceinline__ v8f wmma_f16(v16h a, v16h b, v8f c) {
  return __builtin_amdgcn_wmma_f32_16x16x32_f16(false, a, false, b, (short)0, c, false, false);
}

__global__ __launch_bounds__(256) void eeg_encoder_kernel(
    const float* __restrict__ x,      const float* __restrict__ conv_w,
    const float* __restrict__ pe_w,   const float* __restrict__ wv_all,
    const float* __restrict__ ln1_g,  const float* __restrict__ ln1_b,
    const float* __restrict__ ln2_g,  const float* __restrict__ ln2_b,
    const float* __restrict__ w1_all, const float* __restrict__ b1_all,
    const float* __restrict__ w2_all, const float* __restrict__ b2_all,
    const float* __restrict__ bn_g,   const float* __restrict__ bn_b,
    const float* __restrict__ bn_mean,const float* __restrict__ bn_var,
    const float* __restrict__ fc_w,   float* __restrict__ out)
{
  extern __shared__ char smem[];
  float*    Hs  = (float*)   (smem);
  _Float16* HF  = (_Float16*)(smem + 79360);
  _Float16* WVP = (_Float16*)(smem + 142848);
  _Float16* W1P = (_Float16*)(smem + 148992);
  _Float16* W2P = (_Float16*)(smem + 169472);
  _Float16* T1  = (_Float16*)(smem + 184832);
  float*    SCR = (float*)   (smem + 225792);
  float*    XS  = (float*)   (smem + 142848);   // overlay (stage 0 only)
  float*    CWL = (float*)   (smem + 225792);   // overlay (stage 0 only)
  float*    PEL = (float*)   (smem + 225792 + 2080);
  float*    POOL= (float*)   (smem + 148992);   // overlay (head only)
  float*    PART= (float*)   (smem + 142848);   // overlay (head only)
  float*    LG  = (float*)   (smem + 169472);   // overlay (head only)

  const int b    = blockIdx.x;
  const int tid  = threadIdx.x;
  const int lane = tid & 31;
  const int wave = tid >> 5;

  // ---------------- Stage 0: async-stage x panel into LDS ----------------
  const float* xg = x + (size_t)b * 15000;   // [30][500], 60000 B = 3750 x 16B
  for (int idx = tid; idx < 3750; idx += 256) {
    unsigned lds_off = XS_LDS_OFF + ((unsigned)idx << 4);
    const float* gp = xg + idx * 4;
    asm volatile("global_load_async_to_lds_b128 %0, %1, off"
                 :: "v"(lds_off), "v"(gp) : "memory");
  }
  for (int i = tid; i < 520;  i += 256) CWL[i] = conv_w[i];   // [40][13]
  for (int i = tid; i < 1200; i += 256) PEL[i] = pe_w[i];     // [40][30]
  asm volatile("s_wait_asynccnt 0" ::: "memory");
  __syncthreads();

  // h[t][i] = sum_c pe[i][c] * sum_k x[c][t+k]*cw[i][k]
  for (int idx = tid; idx < 488 * 40; idx += 256) {
    int t = idx / 40, i = idx - t * 40;
    const float* cw = CWL + i * 13;
    const float* pw = PEL + i * 30;
    float acc = 0.f;
    for (int c = 0; c < 30; ++c) {
      const float* xr = XS + c * 500 + t;
      float s = 0.f;
      #pragma unroll
      for (int k = 0; k < 13; ++k) s += xr[k] * cw[k];
      acc += s * pw[c];
    }
    Hs[idx] = acc;
  }
  for (int idx = 488 * 40 + tid; idx < 496 * 40; idx += 256) Hs[idx] = 0.f;
  __syncthreads();

  // ---------------- 12 transformer layers ----------------
  for (int l = 0; l < 12; ++l) {
    const float* Wv  = wv_all + l * 1600;
    const float* W1  = w1_all + l * 6400;
    const float* W2  = w2_all + l * 6400;
    const float* B1  = b1_all + l * 160;
    const float* B2  = b2_all + l * 40;
    const float* G1  = ln1_g + l * 40; const float* Be1 = ln1_b + l * 40;
    const float* G2  = ln2_g + l * 40; const float* Be2 = ln2_b + l * 40;

    // Prefetch next layer's weights into cache while this layer computes.
    if (l < 11) {
      const float* nW1 = w1_all + (l + 1) * 6400;
      const float* nW2 = w2_all + (l + 1) * 6400;
      const float* nWv = wv_all + (l + 1) * 1600;
      for (int i = tid * 16; i < 6400; i += 256 * 16) {
        __builtin_prefetch(nW1 + i, 0, 0);
        __builtin_prefetch(nW2 + i, 0, 0);
      }
      for (int i = tid * 16; i < 1600; i += 256 * 16) __builtin_prefetch(nWv + i, 0, 0);
    }

    // Pack Wv^T as B-fragments: frag = nt*2+ks; element e -> k = ks*32+(lane>>4)*8+(e&7)+(e>>3)*16
    for (int idx = tid; idx < 3072; idx += 256) {
      int e = idx & 15, ln_ = (idx >> 4) & 31, q = idx >> 9;
      int ks = q & 1, nt = q >> 1;
      int n = (ln_ & 15) + nt * 16;
      int k = ks * 32 + ((ln_ >> 4) << 3) + (e & 7) + ((e >> 3) << 4);
      WVP[idx] = (k < 40 && n < 40) ? (_Float16)Wv[n * 40 + k] : (_Float16)0.f;
    }
    // Pack W1^T (out[t][n] = sum_k h[t][k]*W1[n][k]): frag = nt*2+ks, nt 0..9
    for (int idx = tid; idx < 10240; idx += 256) {
      int e = idx & 15, ln_ = (idx >> 4) & 31, q = idx >> 9;
      int ks = q & 1, nt = q >> 1;
      int n = (ln_ & 15) + nt * 16;
      int k = ks * 32 + ((ln_ >> 4) << 3) + (e & 7) + ((e >> 3) << 4);
      W1P[idx] = (k < 40) ? (_Float16)W1[n * 40 + k] : (_Float16)0.f;
    }
    // Pack W2^T: K=160 exact, frag = nt*5+ks
    for (int idx = tid; idx < 7680; idx += 256) {
      int e = idx & 15, ln_ = (idx >> 4) & 31, q = idx >> 9;
      int ks = q % 5, nt = q / 5;
      int n = (ln_ & 15) + nt * 16;
      int k = ks * 32 + ((ln_ >> 4) << 3) + (e & 7) + ((e >> 3) << 4);
      W2P[idx] = (n < 40) ? (_Float16)W2[n * 160 + k] : (_Float16)0.f;
    }
    // Snapshot h -> f16 A panel [496][64] (zero-padded K)
    for (int idx = tid; idx < 496 * 64; idx += 256) {
      int t = idx >> 6, k = idx & 63;
      HF[idx] = (k < 40) ? (_Float16)Hs[t * 40 + k] : (_Float16)0.f;
    }
    __syncthreads();

    // ---- attn = h + h@Wv^T ; h += LN1(attn) ----
    {
      v16h bwv[6];
      #pragma unroll
      for (int q = 0; q < 6; ++q) bwv[q] = load_bp(WVP, q, lane);

      for (int mt = wave; mt < 31; mt += 8) {
        const _Float16* Ab = HF + mt * 16 * 64;
        v16h a0 = load_a(Ab, 64, 0,  lane);
        v16h a1 = load_a(Ab, 64, 32, lane);
        float* S = SCR + wave * 16 * 48;
        #pragma unroll
        for (int nt = 0; nt < 3; ++nt) {
          v8f acc = {0.f,0.f,0.f,0.f,0.f,0.f,0.f,0.f};
          acc = wmma_f16(a0, bwv[nt * 2 + 0], acc);
          acc = wmma_f16(a1, bwv[nt * 2 + 1], acc);
          int n = (lane & 15) + nt * 16;
          int mb = (lane >> 4) << 3;
          #pragma unroll
          for (int r = 0; r < 8; ++r) S[(mb + r) * 48 + n] = acc[r];
        }
        asm volatile("s_wait_dscnt 0" ::: "memory");
        if (lane < 16) {
          int grow = mt * 16 + lane;
          if (grow < 488) {
            float* hr = Hs + grow * 40;
            float a40[40];
            float mean = 0.f;
            #pragma unroll
            for (int i = 0; i < 40; ++i) { a40[i] = hr[i] + S[lane * 48 + i]; mean += a40[i]; }
            mean *= (1.f / 40.f);
            float var = 0.f;
            #pragma unroll
            for (int i = 0; i < 40; ++i) { float d = a40[i] - mean; var += d * d; }
            var *= (1.f / 40.f);
            float rs = rsqrtf(var + 1e-5f);
            #pragma unroll
            for (int i = 0; i < 40; ++i)
              hr[i] += (a40[i] - mean) * rs * G1[i] + Be1[i];
          }
        }
      }
    }
    __syncthreads();

    // Re-snapshot updated h
    for (int idx = tid; idx < 496 * 64; idx += 256) {
      int t = idx >> 6, k = idx & 63;
      HF[idx] = (k < 40) ? (_Float16)Hs[t * 40 + k] : (_Float16)0.f;
    }
    __syncthreads();

    // ---- FFN: f = relu(h@W1^T+b1)@W2^T+b2 ; h += LN2(f) ----
    {
      v16h bw1[20];
      #pragma unroll
      for (int q = 0; q < 20; ++q) bw1[q] = load_bp(W1P, q, lane);

      for (int mt = wave; mt < 31; mt += 8) {
        const _Float16* Ab = HF + mt * 16 * 64;
        _Float16* T = T1 + wave * 16 * 160;
        float*    S = SCR + wave * 16 * 48;
        v16h a0 = load_a(Ab, 64, 0,  lane);
        v16h a1 = load_a(Ab, 64, 32, lane);
        #pragma unroll
        for (int nt1 = 0; nt1 < 10; ++nt1) {
          v8f acc = {0.f,0.f,0.f,0.f,0.f,0.f,0.f,0.f};
          acc = wmma_f16(a0, bw1[nt1 * 2 + 0], acc);
          acc = wmma_f16(a1, bw1[nt1 * 2 + 1], acc);
          int n = (lane & 15) + nt1 * 16;
          int mb = (lane >> 4) << 3;
          float bv = B1[n];
          #pragma unroll
          for (int r = 0; r < 8; ++r) {
            float v = acc[r] + bv;
            T[(mb + r) * 160 + n] = (_Float16)(v > 0.f ? v : 0.f);
          }
        }
        asm volatile("s_wait_dscnt 0" ::: "memory");
        v16h ta[5];
        #pragma unroll
        for (int ks = 0; ks < 5; ++ks) ta[ks] = load_a(T, 160, ks * 32, lane);
        #pragma unroll
        for (int nt2 = 0; nt2 < 3; ++nt2) {
          v8f acc = {0.f,0.f,0.f,0.f,0.f,0.f,0.f,0.f};
          #pragma unroll
          for (int ks = 0; ks < 5; ++ks)
            acc = wmma_f16(ta[ks], load_bp(W2P, nt2 * 5 + ks, lane), acc);
          int n = (lane & 15) + nt2 * 16;
          int mb = (lane >> 4) << 3;
          #pragma unroll
          for (int r = 0; r < 8; ++r) S[(mb + r) * 48 + n] = acc[r];
        }
        asm volatile("s_wait_dscnt 0" ::: "memory");
        if (lane < 16) {
          int grow = mt * 16 + lane;
          if (grow < 488) {
            float* hr = Hs + grow * 40;
            float f40[40];
            float mean = 0.f;
            #pragma unroll
            for (int i = 0; i < 40; ++i) { f40[i] = S[lane * 48 + i] + B2[i]; mean += f40[i]; }
            mean *= (1.f / 40.f);
            float var = 0.f;
            #pragma unroll
            for (int i = 0; i < 40; ++i) { float d = f40[i] - mean; var += d * d; }
            var *= (1.f / 40.f);
            float rs = rsqrtf(var + 1e-5f);
            #pragma unroll
            for (int i = 0; i < 40; ++i)
              hr[i] += (f40[i] - mean) * rs * G2[i] + Be2[i];
          }
        }
      }
    }
    __syncthreads();
  }

  // ---------------- Head: BN -> square -> avgpool(35,7) -> log ----------------
  for (int idx = tid; idx < 2600; idx += 256) {
    int ch = idx / 65, p = idx - ch * 65;
    float sc = bn_g[ch] * rsqrtf(bn_var[ch] + 1e-5f);
    float mn = bn_mean[ch], bb = bn_b[ch];
    float s = 0.f;
    int t0 = p * 7;
    for (int u = 0; u < 35; ++u) {
      float v = (Hs[(t0 + u) * 40 + ch] - mn) * sc + bb;
      s += v * v;
    }
    s *= (1.f / 35.f);
    s = fminf(fmaxf(s, 1e-7f), 1e4f);
    POOL[idx] = logf(s);
  }
  __syncthreads();

  // fc [5,2600] + softmax
  float pj[5] = {0.f, 0.f, 0.f, 0.f, 0.f};
  for (int idx = tid; idx < 2600; idx += 256) {
    float pv = POOL[idx];
    #pragma unroll
    for (int j = 0; j < 5; ++j) pj[j] += pv * fc_w[j * 2600 + idx];
  }
  #pragma unroll
  for (int j = 0; j < 5; ++j) PART[j * 256 + tid] = pj[j];
  __syncthreads();
  if (tid < 5) {
    float s = 0.f;
    for (int t = 0; t < 256; ++t) s += PART[tid * 256 + t];
    LG[tid] = s;
  }
  __syncthreads();
  if (tid == 0) {
    float m = LG[0];
    #pragma unroll
    for (int j = 1; j < 5; ++j) m = fmaxf(m, LG[j]);
    float e[5], se = 0.f;
    #pragma unroll
    for (int j = 0; j < 5; ++j) { e[j] = expf(LG[j] - m); se += e[j]; }
    #pragma unroll
    for (int j = 0; j < 5; ++j) out[b * 5 + j] = e[j] / se;
  }
}

extern "C" void kernel_launch(void* const* d_in, const int* in_sizes, int n_in,
                              void* d_out, int out_size, void* d_ws, size_t ws_size,
                              hipStream_t stream) {
  (void)in_sizes; (void)n_in; (void)out_size; (void)d_ws; (void)ws_size;
  const float* x       = (const float*)d_in[0];
  const float* conv_w  = (const float*)d_in[1];
  const float* pe_w    = (const float*)d_in[2];
  // d_in[3] (wq) and d_in[4] (wk) are dead code in the reference
  const float* wv      = (const float*)d_in[5];
  const float* ln1_g   = (const float*)d_in[6];
  const float* ln1_b   = (const float*)d_in[7];
  const float* ln2_g   = (const float*)d_in[8];
  const float* ln2_b   = (const float*)d_in[9];
  const float* w1      = (const float*)d_in[10];
  const float* b1      = (const float*)d_in[11];
  const float* w2      = (const float*)d_in[12];
  const float* b2      = (const float*)d_in[13];
  const float* bn_g    = (const float*)d_in[14];
  const float* bn_b    = (const float*)d_in[15];
  const float* bn_mean = (const float*)d_in[16];
  const float* bn_var  = (const float*)d_in[17];
  const float* fc_w    = (const float*)d_in[18];
  float* out = (float*)d_out;

  (void)hipFuncSetAttribute((const void*)eeg_encoder_kernel,
                            hipFuncAttributeMaxDynamicSharedMemorySize, SMEM_BYTES);

  eeg_encoder_kernel<<<dim3(256), dim3(256), SMEM_BYTES, stream>>>(
      x, conv_w, pe_w, wv, ln1_g, ln1_b, ln2_g, ln2_b,
      w1, b1, w2, b2, bn_g, bn_b, bn_mean, bn_var, fc_w, out);
}
